// Experts_55155970015686
// MI455X (gfx1250) — matrix operand order repaired and checked by reading, not compiled, not run on hardware
//
#include <hip/hip_runtime.h>
#include <hip/hip_bf16.h>

typedef __bf16 bf16_t;
typedef __attribute__((ext_vector_type(4)))  __bf16 v4bf;
typedef __attribute__((ext_vector_type(8)))  __bf16 v8bf;
typedef __attribute__((ext_vector_type(16))) __bf16 v16bf;
typedef __attribute__((ext_vector_type(8)))  float  v8f;
typedef __attribute__((ext_vector_type(4)))  int    v4i;

#define AS1 __attribute__((address_space(1)))
#define AS3 __attribute__((address_space(3)))

#if defined(__has_builtin)
#if __has_builtin(__builtin_amdgcn_global_load_async_to_lds_b128) && \
    __has_builtin(__builtin_amdgcn_s_wait_asynccnt)
#define HAS_ASYNC_LDS 1
#endif
#endif

__device__ __forceinline__ AS1 v4i* as_global_v4i(const void* p) {
  // global generic addresses are numerically identical to AS1 addresses
  return (AS1 v4i*)(size_t)p;
}
__device__ __forceinline__ AS3 v4i* as_lds_v4i(void* p) {
  return (AS3 v4i*)p;  // addrspacecast generic -> LDS
}

// ---------------------------------------------------------------------------
// 1) LayerNorm over D, fp32 in -> bf16 out.  One block (256 thr) per row.
// ---------------------------------------------------------------------------
__global__ __launch_bounds__(256) void ln_bf16_kernel(
    const float* __restrict__ x, const float* __restrict__ gamma,
    bf16_t* __restrict__ xn, int D) {
  const int row = blockIdx.x;
  const int tid = threadIdx.x;
  const float* xr = x + (size_t)row * D;

  float4 v = *(const float4*)(xr + tid * 4);
  float s  = v.x + v.y + v.z + v.w;
  float ss = v.x * v.x + v.y * v.y + v.z * v.z + v.w * v.w;

#pragma unroll
  for (int off = 16; off > 0; off >>= 1) {
    s  += __shfl_xor(s,  off, 32);
    ss += __shfl_xor(ss, off, 32);
  }

  __shared__ float rs[8], rss[8];
  if ((tid & 31) == 0) { rs[tid >> 5] = s; rss[tid >> 5] = ss; }
  __syncthreads();
  float tot = 0.f, tot2 = 0.f;
#pragma unroll
  for (int i = 0; i < 8; ++i) { tot += rs[i]; tot2 += rss[i]; }

  const float invD = 1.0f / (float)D;
  const float mean = tot * invD;
  const float var  = tot2 * invD - mean * mean;
  const float rstd = rsqrtf(var + 1e-5f);

  float4 g = *(const float4*)(gamma + tid * 4);
  v4bf o;
  o[0] = (bf16_t)((v.x - mean) * rstd * g.x);
  o[1] = (bf16_t)((v.y - mean) * rstd * g.y);
  o[2] = (bf16_t)((v.z - mean) * rstd * g.z);
  o[3] = (bf16_t)((v.w - mean) * rstd * g.w);
  *(v4bf*)(xn + (size_t)row * D + tid * 4) = o;
}

// ---------------------------------------------------------------------------
// 2) Batched transpose + fp32->bf16 convert: in [Z][R][C] f32 -> out [Z][C][R] bf16
// ---------------------------------------------------------------------------
__global__ __launch_bounds__(256) void tconv_kernel(
    const float* __restrict__ in, bf16_t* __restrict__ out, int R, int C) {
  __shared__ float tile[32][33];
  const int z = blockIdx.z;
  const float* ib = in + (size_t)z * R * C;
  bf16_t* ob = out + (size_t)z * R * C;
  const int r0 = blockIdx.y * 32, c0 = blockIdx.x * 32;
  const int tx = threadIdx.x & 31, ty = threadIdx.x >> 5;

#pragma unroll
  for (int s = 0; s < 4; ++s) {
    int r = ty + s * 8;
    tile[r][tx] = ib[(size_t)(r0 + r) * C + (c0 + tx)];
  }
  __syncthreads();
#pragma unroll
  for (int s = 0; s < 4; ++s) {
    int r = ty + s * 8;
    ob[(size_t)(c0 + r) * R + (r0 + tx)] = (bf16_t)tile[tx][r];
  }
}

// ---------------------------------------------------------------------------
// 3) bf16 WMMA GEMM:  C[z] (MxNn) = A[z] (MxK) * Bt[z%E]^T   (Bt stored NnxK)
//    Block tile 128x256x32, 8 waves (2x4), wave tile 64x64 = 4x4 WMMA accums.
//    Double-buffered LDS, async global->LDS staging when available.
// ---------------------------------------------------------------------------
template <bool GELU, typename OutT>
__global__ __launch_bounds__(256) void gemm_wmma_bf16(
    const bf16_t* __restrict__ A, const bf16_t* __restrict__ Bt,
    OutT* __restrict__ C, int M, int Nn, int K, int Emod) {
  constexpr int BM = 128, BN = 256, BK = 32;
  constexpr int LDA = BK + 8;  // 40 halves = 80B row stride (bank spread)

  __shared__ __align__(16) bf16_t As[2][BM * LDA];
  __shared__ __align__(16) bf16_t Bs[2][BN * LDA];

  const int z = blockIdx.z;
  const int e = z % Emod;
  const bf16_t* Ab = A  + (size_t)z * M  * K;
  const bf16_t* Bb = Bt + (size_t)e * Nn * K;
  OutT*         Cb = C  + (size_t)z * M  * Nn;

  const int tileM = blockIdx.x * BM;
  const int tileN = blockIdx.y * BN;
  const int tid   = threadIdx.x;
  const int wave  = tid >> 5;
  const int lane  = tid & 31;
  const int wm    = (wave & 1) * 64;   // 2 waves along M
  const int wn    = (wave >> 1) * 64;  // 4 waves along N
  const int row16 = lane & 15;
  const int hi    = lane >> 4;

  v8f acc[4][4];
#pragma unroll
  for (int i = 0; i < 4; ++i)
#pragma unroll
    for (int j = 0; j < 4; ++j)
#pragma unroll
      for (int r = 0; r < 8; ++r) acc[i][j][r] = 0.0f;

  // A tile = 512 16B chunks (2/thread); B tile = 1024 16B chunks (4/thread)
#if defined(HAS_ASYNC_LDS)
  auto issue_tile = [&](int k0, int buf) {
#pragma unroll
    for (int s = 0; s < 2; ++s) {
      int c = tid + s * 256, r = c >> 2, col = (c & 3) << 3;
      __builtin_amdgcn_global_load_async_to_lds_b128(
          as_global_v4i(Ab + (size_t)(tileM + r) * K + k0 + col),
          as_lds_v4i(&As[buf][r * LDA + col]), 0, 0);
    }
#pragma unroll
    for (int s = 0; s < 4; ++s) {
      int c = tid + s * 256, r = c >> 2, col = (c & 3) << 3;
      __builtin_amdgcn_global_load_async_to_lds_b128(
          as_global_v4i(Bb + (size_t)(tileN + r) * K + k0 + col),
          as_lds_v4i(&Bs[buf][r * LDA + col]), 0, 0);
    }
  };
#else
  auto g_addr_a = [&](int k0, int s) -> const v8bf* {
    int c = tid + s * 256, r = c >> 2, col = (c & 3) << 3;
    return (const v8bf*)(Ab + (size_t)(tileM + r) * K + k0 + col);
  };
  auto g_addr_b = [&](int k0, int s) -> const v8bf* {
    int c = tid + s * 256, r = c >> 2, col = (c & 3) << 3;
    return (const v8bf*)(Bb + (size_t)(tileN + r) * K + k0 + col);
  };
  auto s_store = [&](int buf, const v8bf* pa, const v8bf* pb) {
#pragma unroll
    for (int s = 0; s < 2; ++s) {
      int c = tid + s * 256, r = c >> 2, col = (c & 3) << 3;
      *(v8bf*)&As[buf][r * LDA + col] = pa[s];
    }
#pragma unroll
    for (int s = 0; s < 4; ++s) {
      int c = tid + s * 256, r = c >> 2, col = (c & 3) << 3;
      *(v8bf*)&Bs[buf][r * LDA + col] = pb[s];
    }
  };
#endif

  // ---- prologue: fill buffer 0 ----
#if defined(HAS_ASYNC_LDS)
  issue_tile(0, 0);
  __builtin_amdgcn_s_wait_asynccnt(0);
#else
  {
    v8bf pa[2], pb[4];
#pragma unroll
    for (int s = 0; s < 2; ++s) pa[s] = *g_addr_a(0, s);
#pragma unroll
    for (int s = 0; s < 4; ++s) pb[s] = *g_addr_b(0, s);
    s_store(0, pa, pb);
  }
#endif
  __syncthreads();

  const int nk = K / BK;
  for (int kt = 0; kt < nk; ++kt) {
    const int buf = kt & 1;
    const bool pf = (kt + 1 < nk);

#if defined(HAS_ASYNC_LDS)
    if (pf) issue_tile((kt + 1) * BK, buf ^ 1);
#else
    v8bf pa[2], pb[4];
    if (pf) {
      int k0 = (kt + 1) * BK;
#pragma unroll
      for (int s = 0; s < 2; ++s) pa[s] = *g_addr_a(k0, s);
#pragma unroll
      for (int s = 0; s < 4; ++s) pb[s] = *g_addr_b(k0, s);
    }
#endif

    // A fragments: lane holds row (lane&15); K = hi*8 + {0..7} and hi*8+16+{0..7}
    v16bf af[4];
#pragma unroll
    for (int i = 0; i < 4; ++i) {
      const bf16_t* p = &As[buf][(wm + i * 16 + row16) * LDA + hi * 8];
      v8bf lo  = *(const v8bf*)(p);
      v8bf hi8 = *(const v8bf*)(p + 16);
      af[i] = __builtin_shufflevector(lo, hi8, 0, 1, 2, 3, 4, 5, 6, 7,
                                      8, 9, 10, 11, 12, 13, 14, 15);
    }
    // B fragments: lane holds column (lane&15); K = hi*8 + {0..7} and hi*8+16+{0..7}
    v16bf bfr[4];
#pragma unroll
    for (int j = 0; j < 4; ++j) {
      const bf16_t* p = &Bs[buf][(wn + j * 16 + row16) * LDA + hi * 8];
      v8bf lo  = *(const v8bf*)(p);
      v8bf hi8 = *(const v8bf*)(p + 16);
      bfr[j] = __builtin_shufflevector(lo, hi8, 0, 1, 2, 3, 4, 5, 6, 7,
                                       8, 9, 10, 11, 12, 13, 14, 15);
    }

#pragma unroll
    for (int i = 0; i < 4; ++i)
#pragma unroll
      for (int j = 0; j < 4; ++j)
        acc[i][j] = __builtin_amdgcn_wmma_f32_16x16x32_bf16(
            false, af[i], false, bfr[j], (short)0, acc[i][j], false, false);

#if defined(HAS_ASYNC_LDS)
    __builtin_amdgcn_s_wait_asynccnt(0);
#else
    if (pf) s_store(buf ^ 1, pa, pb);
#endif
    __syncthreads();
  }

  // epilogue: acc elem r of lane -> C row (tile row + r + hi*8), col (lane&15)
#pragma unroll
  for (int i = 0; i < 4; ++i) {
#pragma unroll
    for (int j = 0; j < 4; ++j) {
      const int gcol  = tileN + wn + j * 16 + row16;
      const int grow0 = tileM + wm + i * 16 + hi * 8;
#pragma unroll
      for (int r = 0; r < 8; ++r) {
        float v = acc[i][j][r];
        if (GELU) v = 0.5f * v * (1.0f + erff(v * 0.70710678118654752f));
        Cb[(size_t)(grow0 + r) * Nn + gcol] = (OutT)v;
      }
    }
  }
}

// ---------------------------------------------------------------------------
extern "C" void kernel_launch(void* const* d_in, const int* in_sizes, int n_in,
                              void* d_out, int out_size, void* d_ws, size_t ws_size,
                              hipStream_t stream) {
  constexpr int B = 2, E = 8, N = 2048, D = 1024, H = 4096;
  const float* x     = (const float*)d_in[0];
  const float* gamma = (const float*)d_in[1];
  const float* w1    = (const float*)d_in[2];  // [E][D][H]
  const float* w2    = (const float*)d_in[3];  // [E][H][D]
  float* out = (float*)d_out;                  // [B][E][N][D]

  char* ws = (char*)d_ws;
  size_t off = 0;
  bf16_t* xn  = (bf16_t*)(ws + off); off += (size_t)B * E * N * D * sizeof(bf16_t);
  bf16_t* w1t = (bf16_t*)(ws + off); off += (size_t)E * H * D * sizeof(bf16_t);
  bf16_t* w2t = (bf16_t*)(ws + off); off += (size_t)E * D * H * sizeof(bf16_t);
  bf16_t* hid = (bf16_t*)(ws + off);           // [B*E][N][H]

  ln_bf16_kernel<<<B * E * N, 256, 0, stream>>>(x, gamma, xn, D);
  tconv_kernel<<<dim3(H / 32, D / 32, E), 256, 0, stream>>>(w1, w1t, D, H);
  tconv_kernel<<<dim3(D / 32, H / 32, E), 256, 0, stream>>>(w2, w2t, H, D);
  // hidden = gelu(xn @ w1)     M=N, Nn=H, K=D
  gemm_wmma_bf16<true, bf16_t>
      <<<dim3(N / 128, H / 256, B * E), 256, 0, stream>>>(xn, w1t, hid, N, H, D, E);
  // out = hidden @ w2          M=N, Nn=D, K=H
  gemm_wmma_bf16<false, float>
      <<<dim3(N / 128, D / 256, B * E), 256, 0, stream>>>(hid, w2t, out, N, D, H, E);
}
